// TI_GATWithEdgeAttrNodeProj_39479339384966
// MI455X (gfx1250) — matrix operand order repaired and checked by reading, not compiled, not run on hardware
//
#include <hip/hip_runtime.h>

// ---------------- static problem config (matches reference) ----------------
#define NN      32000            // nodes
#define FIN     128              // input features
#define HCDIM   64               // heads*channels
#define NHEAD   4
#define CHEAD   16
#define EEDGES  512000           // edges before self loops
#define EFULL   (EEDGES + NN)    // with self loops = 544000
#define EDIM    16               // edge_attr dim
#define PROJ    8
#define OUTD    256
#define NGRAPH  32
#define KFC     8000             // 1000*PROJ
#define SLOPE   0.2f

typedef __attribute__((ext_vector_type(16))) _Float16 v16h;
typedef __attribute__((ext_vector_type(8)))  float    v8f;

// ------------- order-preserving float<->uint encoding for atomicMax --------
__device__ __forceinline__ unsigned fenc(float f) {
    unsigned u = __float_as_uint(f);
    return (u & 0x80000000u) ? ~u : (u | 0x80000000u);
}
__device__ __forceinline__ float fdec(unsigned u) {
    unsigned v = (u & 0x80000000u) ? (u & 0x7FFFFFFFu) : ~u;
    return __uint_as_float(v);
}
#define ENC_NEG_INF 0x007FFFFFu   // fenc(-inf)

// ------------------------ WMMA tile loaders (ISA 7.12.2) -------------------
// A 16x32 f16: lanes 0-15 = rows, lane group selects K {0-7,16-23} vs {8-15,24-31}
__device__ __forceinline__ v16h load_tileA(const _Float16* __restrict__ X,
                                           int row0, int k0, int ld) {
    int lane = threadIdx.x & 31;
    int r    = row0 + (lane & 15);
    int kq   = (lane >> 4) * 8;
    const _Float16* p = X + (long)r * ld + k0 + kq;
    v16h a;
#pragma unroll
    for (int t = 0; t < 8; ++t) { a[t] = p[t]; a[8 + t] = p[16 + t]; }
    return a;
}
// B operand pre-swizzled: tile(kt,nt) -> 32 lanes x 16 contiguous halves (32B/lane)
__device__ __forceinline__ v16h load_tileBp(const _Float16* __restrict__ Bp,
                                            int kt, int nt, int NT) {
    int lane = threadIdx.x & 31;
    const _Float16* p = Bp + (((long)(kt * NT + nt) * 32 + lane) << 4);
    v16h b;
#pragma unroll
    for (int t = 0; t < 16; ++t) b[t] = p[t];
    return b;
}

// ------------------------------- utility kernels ---------------------------
__global__ void k_cast_f16(const float* __restrict__ in, _Float16* __restrict__ out, int n) {
    int i = blockIdx.x * blockDim.x + threadIdx.x;
    if (i < n) out[i] = (_Float16)in[i];
}
// pack row-major W[K,Ncols] f32 into WMMA B-operand tile layout (f16).
// out index i = ((kt*NT + nt)*32 + lane)*16 + t ;
// source: K = kt*32 + (lane>>4)*8 + (t>>3)*16 + (t&7) , N = nt*16 + (lane&15)
__global__ void k_packB(const float* __restrict__ W, _Float16* __restrict__ Bp,
                        int K, int Ncols) {
    int i = blockIdx.x * blockDim.x + threadIdx.x;
    if (i >= K * Ncols) return;
    int t    = i & 15;
    int lane = (i >> 4) & 31;
    int tile = i >> 9;
    int NT   = Ncols >> 4;
    int kt   = tile / NT, nt = tile % NT;
    int k    = kt * 32 + (lane >> 4) * 8 + (t >> 3) * 16 + (t & 7);
    int n    = nt * 16 + (lane & 15);
    Bp[i] = (_Float16)W[(long)k * Ncols + n];
}
__global__ void k_fill_f32(float* __restrict__ p, float v, int n) {
    int i = blockIdx.x * blockDim.x + threadIdx.x;
    if (i < n) p[i] = v;
}
__global__ void k_fill_u32(unsigned* __restrict__ p, unsigned v, int n) {
    int i = blockIdx.x * blockDim.x + threadIdx.x;
    if (i < n) p[i] = v;
}
__global__ void k_bias_init(float* __restrict__ h, const float* __restrict__ b, int n) {
    int i = blockIdx.x * blockDim.x + threadIdx.x;
    if (i < n) h[i] = b[i & (HCDIM - 1)];
}

// ----------------- self-loop attr: deg + per-dst mean of edge_attr ---------
__global__ void k_degsum(const int* __restrict__ dst, const float* __restrict__ ea,
                         float* __restrict__ la, float* __restrict__ deg) {
    int i = blockIdx.x * blockDim.x + threadIdx.x;
    if (i >= EEDGES * EDIM) return;
    int e = i >> 4, k = i & 15;
    int d = dst[e];
    atomicAdd(&la[d * EDIM + k], ea[i]);
    if (k == 0) atomicAdd(&deg[d], 1.0f);
}
__global__ void k_lanorm(float* __restrict__ la, const float* __restrict__ deg) {
    int i = blockIdx.x * blockDim.x + threadIdx.x;
    if (i >= NN * EDIM) return;
    la[i] /= fmaxf(deg[i >> 4], 1.0f);
}

// -------- pre-contract edge projection:  ve[k][h] = sum_c We[k,h*16+c]*ae[h,c]
__global__ void k_ve(const float* __restrict__ We, const float* __restrict__ ae,
                     float* __restrict__ ve) {
    int i = threadIdx.x;
    if (i >= EDIM * NHEAD) return;
    int k = i / NHEAD, h = i % NHEAD;
    float s = 0.f;
#pragma unroll
    for (int c = 0; c < CHEAD; ++c) s += We[k * HCDIM + h * CHEAD + c] * ae[h * CHEAD + c];
    ve[i] = s;
}
// al_e[e,h] = eattr_f[e,:] @ ve[:,h]   (self-loop rows come from la)
__global__ void k_ale(const float* __restrict__ ea, const float* __restrict__ la,
                      const float* __restrict__ ve, float* __restrict__ ale) {
    int e = blockIdx.x * blockDim.x + threadIdx.x;
    if (e >= EFULL) return;
    const float* p = (e < EEDGES) ? (ea + (long)e * EDIM) : (la + (long)(e - EEDGES) * EDIM);
    float s0 = 0.f, s1 = 0.f, s2 = 0.f, s3 = 0.f;
#pragma unroll
    for (int k = 0; k < EDIM; ++k) {
        float v = p[k];
        s0 += v * ve[k * 4 + 0]; s1 += v * ve[k * 4 + 1];
        s2 += v * ve[k * 4 + 2]; s3 += v * ve[k * 4 + 3];
    }
    ale[e * 4 + 0] = s0; ale[e * 4 + 1] = s1; ale[e * 4 + 2] = s2; ale[e * 4 + 3] = s3;
}

// ---------------- WMMA GEMM: Y[M,64] = Xh[M,K] @ Bp(K,64) ------------------
__global__ void k_gemm_n64(const _Float16* __restrict__ X, const _Float16* __restrict__ Bp,
                           float* __restrict__ Y, int M, int K) {
    int wave = blockIdx.x * (blockDim.x >> 5) + (threadIdx.x >> 5);
    int row0 = wave << 4;
    if (row0 >= M) return;                 // uniform per wave: EXEC all-ones in WMMA
    v8f z = {0.f, 0.f, 0.f, 0.f, 0.f, 0.f, 0.f, 0.f};
    v8f acc0 = z, acc1 = z, acc2 = z, acc3 = z;
    int KT = K >> 5;
    for (int kt = 0; kt < KT; ++kt) {
        v16h a  = load_tileA(X, row0, kt * 32, K);
        v16h b0 = load_tileBp(Bp, kt, 0, 4);
        v16h b1 = load_tileBp(Bp, kt, 1, 4);
        v16h b2 = load_tileBp(Bp, kt, 2, 4);
        v16h b3 = load_tileBp(Bp, kt, 3, 4);
        acc0 = __builtin_amdgcn_wmma_f32_16x16x32_f16(false, a, false, b0, (short)0, acc0, false, false);
        acc1 = __builtin_amdgcn_wmma_f32_16x16x32_f16(false, a, false, b1, (short)0, acc1, false, false);
        acc2 = __builtin_amdgcn_wmma_f32_16x16x32_f16(false, a, false, b2, (short)0, acc2, false, false);
        acc3 = __builtin_amdgcn_wmma_f32_16x16x32_f16(false, a, false, b3, (short)0, acc3, false, false);
    }
    int lane = threadIdx.x & 31;
    int n = lane & 15, mb = (lane >> 4) * 8;
#pragma unroll
    for (int i = 0; i < 8; ++i) {
        long r = (long)(row0 + mb + i) * HCDIM;
        Y[r +  0 + n] = acc0[i];
        Y[r + 16 + n] = acc1[i];
        Y[r + 32 + n] = acc2[i];
        Y[r + 48 + n] = acc3[i];
    }
}

// al_s / al_d per node from xl
__global__ void k_alsd(const float* __restrict__ xl, const float* __restrict__ as_,
                       const float* __restrict__ ad_, float* __restrict__ als,
                       float* __restrict__ ald) {
    int n = blockIdx.x * blockDim.x + threadIdx.x;
    if (n >= NN) return;
#pragma unroll
    for (int h = 0; h < NHEAD; ++h) {
        float ss = 0.f, sd = 0.f;
#pragma unroll
        for (int c = 0; c < CHEAD; ++c) {
            float v = xl[(long)n * HCDIM + h * CHEAD + c];
            ss += v * as_[h * CHEAD + c];
            sd += v * ad_[h * CHEAD + c];
        }
        als[n * 4 + h] = ss; ald[n * 4 + h] = sd;
    }
}

// logits + segment max (atomicMax on order-preserving encoding)
__global__ void k_logmax(const int* __restrict__ src, const int* __restrict__ dst,
                         const float* __restrict__ als, const float* __restrict__ ald,
                         const float* __restrict__ ale, float* __restrict__ logits,
                         unsigned* __restrict__ mx) {
    int e = blockIdx.x * blockDim.x + threadIdx.x;
    if (e >= EFULL) return;
    int s, d;
    if (e < EEDGES) { s = src[e]; d = dst[e]; } else { s = d = e - EEDGES; }
#pragma unroll
    for (int h = 0; h < NHEAD; ++h) {
        float l = als[s * 4 + h] + ald[d * 4 + h] + ale[e * 4 + h];
        l = (l > 0.f) ? l : SLOPE * l;               // leaky_relu
        logits[e * 4 + h] = l;
        atomicMax(&mx[d * 4 + h], fenc(l));
    }
}

// p = exp(logit - mx[dst]); denom += p   (p overwrites logits buffer)
__global__ void k_expsum(const int* __restrict__ dst, float* __restrict__ logits,
                         const unsigned* __restrict__ mx, float* __restrict__ denom) {
    int e = blockIdx.x * blockDim.x + threadIdx.x;
    if (e >= EFULL) return;
    int d = (e < EEDGES) ? dst[e] : (e - EEDGES);
#pragma unroll
    for (int h = 0; h < NHEAD; ++h) {
        float p = expf(logits[e * 4 + h] - fdec(mx[d * 4 + h]));
        logits[e * 4 + h] = p;
        atomicAdd(&denom[d * 4 + h], p);
    }
}

// message scatter: hout[dst] += xl[src] * alpha  (2 features / thread)
__global__ void k_scatter(const int* __restrict__ src, const int* __restrict__ dst,
                          const float* __restrict__ p, const float* __restrict__ denom,
                          const float* __restrict__ xl, float* __restrict__ hout) {
    int idx = blockIdx.x * blockDim.x + threadIdx.x;
    int e = idx >> 5;
    if (e >= EFULL) return;
    int j = idx & 31;
    int s, d;
    if (e < EEDGES) { s = src[e]; d = dst[e]; } else { s = d = e - EEDGES; }
    int f = j * 2;
    int h = f >> 4;
    float alpha = p[e * 4 + h] / denom[d * 4 + h];
    long sb = (long)s * HCDIM + f, db = (long)d * HCDIM + f;
    atomicAdd(&hout[db],     xl[sb]     * alpha);
    atomicAdd(&hout[db + 1], xl[sb + 1] * alpha);
}

// relu in place + f16 copy for next layer's WMMA
__global__ void k_relucast(float* __restrict__ h, _Float16* __restrict__ hh, int n) {
    int i = blockIdx.x * blockDim.x + threadIdx.x;
    if (i >= n) return;
    float v = fmaxf(h[i], 0.f);
    h[i] = v;
    hh[i] = (_Float16)v;
}

// node projection 64->8, relu, straight to f16 (feeds FC WMMA)
__global__ void k_proj(const float* __restrict__ h, const float* __restrict__ Wp,
                       const float* __restrict__ bp, _Float16* __restrict__ hph) {
    int n = blockIdx.x * blockDim.x + threadIdx.x;
    if (n >= NN) return;
    float acc[PROJ];
#pragma unroll
    for (int p = 0; p < PROJ; ++p) acc[p] = bp[p];
    for (int j = 0; j < HCDIM; ++j) {
        float v = h[(long)n * HCDIM + j];
#pragma unroll
        for (int p = 0; p < PROJ; ++p) acc[p] += v * Wp[j * PROJ + p];
    }
#pragma unroll
    for (int p = 0; p < PROJ; ++p) hph[(long)n * PROJ + p] = (_Float16)fmaxf(acc[p], 0.f);
}

// final FC: [32,8000] @ Bp(8000,256) + bo -> d_out, WMMA, one wave / 16x16 tile
__global__ void k_fc(const _Float16* __restrict__ A, const _Float16* __restrict__ Bp,
                     const float* __restrict__ bias, float* __restrict__ Y) {
    int wave = blockIdx.x * (blockDim.x >> 5) + (threadIdx.x >> 5);
    if (wave >= (NGRAPH / 16) * (OUTD / 16)) return;
    int mt = wave >> 4, nt = wave & 15;
    v8f acc = {0.f, 0.f, 0.f, 0.f, 0.f, 0.f, 0.f, 0.f};
    for (int kt = 0; kt < KFC / 32; ++kt) {
        v16h a = load_tileA(A, mt * 16, kt * 32, KFC);
        v16h b = load_tileBp(Bp, kt, nt, OUTD / 16);
        acc = __builtin_amdgcn_wmma_f32_16x16x32_f16(false, a, false, b, (short)0, acc, false, false);
    }
    int lane = threadIdx.x & 31;
    int n = nt * 16 + (lane & 15), mb = (lane >> 4) * 8;
#pragma unroll
    for (int i = 0; i < 8; ++i)
        Y[(long)(mt * 16 + mb + i) * OUTD + n] = acc[i] + bias[n];
}

// --------------------------------- host side -------------------------------
static inline unsigned gridFor(long n, int blk) { return (unsigned)((n + blk - 1) / blk); }

extern "C" void kernel_launch(void* const* d_in, const int* in_sizes, int n_in,
                              void* d_out, int out_size, void* d_ws, size_t ws_size,
                              hipStream_t stream) {
    (void)in_sizes; (void)n_in; (void)out_size; (void)ws_size;
    const float* x    = (const float*)d_in[0];
    const float* ea   = (const float*)d_in[1];
    const float* W0   = (const float*)d_in[2];
    const float* a_s0 = (const float*)d_in[3];
    const float* a_d0 = (const float*)d_in[4];
    const float* We0  = (const float*)d_in[5];
    const float* ae0  = (const float*)d_in[6];
    const float* b0   = (const float*)d_in[7];
    const float* Wg   = (const float*)d_in[8];
    const float* a_sg = (const float*)d_in[9];
    const float* a_dg = (const float*)d_in[10];
    const float* Weg  = (const float*)d_in[11];
    const float* aeg  = (const float*)d_in[12];
    const float* bg   = (const float*)d_in[13];
    const float* Wp   = (const float*)d_in[14];
    const float* bp   = (const float*)d_in[15];
    const float* Wo   = (const float*)d_in[16];
    const float* bo   = (const float*)d_in[17];
    const int*   eidx = (const int*)d_in[18];
    const int*   src  = eidx;
    const int*   dst  = eidx + EEDGES;
    float*       out  = (float*)d_out;

    // ---- bump allocator in workspace (256B aligned) ----
    char*  base = (char*)d_ws;
    size_t off  = 0;
    auto alloc = [&](size_t bytes) -> void* {
        void* p = base + off;
        off += (bytes + 255) & ~(size_t)255;
        return p;
    };
    _Float16* xh   = (_Float16*)alloc((size_t)NN * FIN * 2);
    _Float16* W0p  = (_Float16*)alloc((size_t)FIN * HCDIM * 2);        // packed B
    _Float16* Wgp  = (_Float16*)alloc((size_t)2 * HCDIM * HCDIM * 2);  // packed B x2
    _Float16* Wop  = (_Float16*)alloc((size_t)KFC * OUTD * 2);         // packed B
    float*    deg  = (float*)alloc((size_t)NN * 4);
    float*    la   = (float*)alloc((size_t)NN * EDIM * 4);
    float*    ve   = (float*)alloc((size_t)EDIM * NHEAD * 4);
    float*    ale  = (float*)alloc((size_t)EFULL * NHEAD * 4);
    float*    lg   = (float*)alloc((size_t)EFULL * NHEAD * 4);   // logits -> p
    unsigned* mx   = (unsigned*)alloc((size_t)NN * NHEAD * 4);
    float*    den  = (float*)alloc((size_t)NN * NHEAD * 4);
    float*    als  = (float*)alloc((size_t)NN * NHEAD * 4);
    float*    ald  = (float*)alloc((size_t)NN * NHEAD * 4);
    float*    xl   = (float*)alloc((size_t)NN * HCDIM * 4);
    float*    hA   = (float*)alloc((size_t)NN * HCDIM * 4);
    float*    hB   = (float*)alloc((size_t)NN * HCDIM * 4);
    _Float16* hh   = (_Float16*)alloc((size_t)NN * HCDIM * 2);
    _Float16* hph  = (_Float16*)alloc((size_t)NN * PROJ * 2);

    const int BLK = 256;

    // ---- input cast + weight packing into WMMA B-operand layout ----
    k_cast_f16<<<gridFor((long)NN * FIN, BLK), BLK, 0, stream>>>(x, xh, NN * FIN);
    k_packB<<<gridFor(FIN * HCDIM, BLK), BLK, 0, stream>>>(W0, W0p, FIN, HCDIM);
    k_packB<<<gridFor(HCDIM * HCDIM, BLK), BLK, 0, stream>>>(Wg, Wgp, HCDIM, HCDIM);
    k_packB<<<gridFor(HCDIM * HCDIM, BLK), BLK, 0, stream>>>(Wg + HCDIM * HCDIM,
                                                             Wgp + HCDIM * HCDIM, HCDIM, HCDIM);
    k_packB<<<gridFor((long)KFC * OUTD, BLK), BLK, 0, stream>>>(Wo, Wop, KFC, OUTD);

    // ---- self-loop edge_attr (per-dst mean) ----
    k_fill_f32<<<gridFor(NN, BLK), BLK, 0, stream>>>(deg, 0.f, NN);
    k_fill_f32<<<gridFor((long)NN * EDIM, BLK), BLK, 0, stream>>>(la, 0.f, NN * EDIM);
    k_degsum<<<gridFor((long)EEDGES * EDIM, BLK), BLK, 0, stream>>>(dst, ea, la, deg);
    k_lanorm<<<gridFor((long)NN * EDIM, BLK), BLK, 0, stream>>>(la, deg);

    // ---- one GAT layer ----
    auto runLayer = [&](const _Float16* Xh, int K, const _Float16* Bp,
                        const float* as_, const float* ad_,
                        const float* We_, const float* ae_,
                        const float* bias, float* Hout) {
        k_ve<<<1, 64, 0, stream>>>(We_, ae_, ve);
        k_ale<<<gridFor(EFULL, BLK), BLK, 0, stream>>>(ea, la, ve, ale);
        // xl = Xh @ W (WMMA)
        k_gemm_n64<<<gridFor((long)(NN / 16) * 32, BLK), BLK, 0, stream>>>(Xh, Bp, xl, NN, K);
        k_alsd<<<gridFor(NN, BLK), BLK, 0, stream>>>(xl, as_, ad_, als, ald);
        k_fill_u32<<<gridFor((long)NN * NHEAD, BLK), BLK, 0, stream>>>(mx, ENC_NEG_INF, NN * NHEAD);
        k_fill_f32<<<gridFor((long)NN * NHEAD, BLK), BLK, 0, stream>>>(den, 0.f, NN * NHEAD);
        k_logmax<<<gridFor(EFULL, BLK), BLK, 0, stream>>>(src, dst, als, ald, ale, lg, mx);
        k_expsum<<<gridFor(EFULL, BLK), BLK, 0, stream>>>(dst, lg, mx, den);
        k_bias_init<<<gridFor((long)NN * HCDIM, BLK), BLK, 0, stream>>>(Hout, bias, NN * HCDIM);
        k_scatter<<<gridFor((long)EFULL * 32, BLK), BLK, 0, stream>>>(src, dst, lg, den, xl, Hout);
        k_relucast<<<gridFor((long)NN * HCDIM, BLK), BLK, 0, stream>>>(Hout, hh, NN * HCDIM);
    };

    // layer 0: 128 -> 64
    runLayer(xh, FIN, W0p, a_s0, a_d0, We0, ae0, b0, hA);
    // layer 1: 64 -> 64
    runLayer(hh, HCDIM, Wgp, a_sg, a_dg, Weg, aeg, bg, hB);
    // layer 2: 64 -> 64
    runLayer(hh, HCDIM, Wgp + HCDIM * HCDIM, a_sg + NHEAD * CHEAD, a_dg + NHEAD * CHEAD,
             Weg + EDIM * HCDIM, aeg + NHEAD * CHEAD, bg + HCDIM, hA);

    // node projection + final FC (WMMA)
    k_proj<<<gridFor(NN, BLK), BLK, 0, stream>>>(hA, Wp, bp, hph);
    k_fc<<<gridFor((long)((NGRAPH / 16) * (OUTD / 16)) * 32, BLK), BLK, 0, stream>>>(hph, Wop, bo, out);
}
